// MixtureOfExperts_57415122813659
// MI455X (gfx1250) — matrix operand order repaired and checked by reading, not compiled, not run on hardware
//
#include <hip/hip_runtime.h>
#include <stdint.h>
#include <stddef.h>

// ---------------- problem constants ----------------
#define N_TOK 8192
#define DIM   1024
#define HID   4096
#define NEXP  8
#define TOPK  2
#define NASSIGN (N_TOK * TOPK)        // 16384 token-expert assignments

// ---------------- GEMM tiling ----------------
#define TM 128
#define TN 128
#define TK 32
#define LDSS 40                        // LDS row stride (halfs) = 80B -> conflict-free b128 reads
#define TILE_HALFS (TM * LDSS)         // one LDS buffer (A or B) in halfs
#define MAXROWS   (NASSIGN + NEXP * TM)       // 17408 (per-expert padding to tile multiples)
#define MAX_MTILES (NASSIGN / TM + NEXP)      // 136

// meta[] layout (ints): 0..7 counts, 8..15 cursor, 16..23 rowStart(padded),
// 24..31 mTileStart, 32 totalMTiles, 33 totalPaddedRows
#define META_INTS 64

typedef __attribute__((ext_vector_type(16))) __bf16    bf16x16;
typedef __attribute__((ext_vector_type(8)))  float     floatx8;
typedef __attribute__((ext_vector_type(4)))  uint32_t  u32x4;
typedef __attribute__((ext_vector_type(2)))  uint32_t  u32x2;
typedef __attribute__((ext_vector_type(4)))  float     f32x4;

struct Frag32 { u32x4 a, b; };

__device__ __forceinline__ uint16_t f32_bf16(float f) {
  union { float f; uint32_t u; } x; x.f = f;
  uint32_t u = x.u;
  u += 0x7FFFu + ((u >> 16) & 1u);     // round-to-nearest-even
  return (uint16_t)(u >> 16);
}

__device__ __forceinline__ bf16x16 lds_frag(const uint16_t* p0, const uint16_t* p1) {
  Frag32 f;
  f.a = *(const u32x4*)p0;
  f.b = *(const u32x4*)p1;
  return __builtin_bit_cast(bf16x16, f);
}

// ---- CDNA5 async global->LDS copy (ASYNCcnt-tracked, no VGPR staging) ----
// Generic shared-memory addresses carry the LDS byte offset in their low 32
// bits (flat LDS aperture), so truncation gives the
// per-lane LDS address VGPR the instruction expects.
__device__ __forceinline__ void async_ld_b128(const void* gptr, void* lptr) {
  asm volatile("global_load_async_to_lds_b128 %0, %1, off"
               :: "v"((uint32_t)(uintptr_t)lptr), "v"(gptr)
               : "memory");
}
__device__ __forceinline__ void wait_async_le0() {
  asm volatile("s_wait_asynccnt 0x0" ::: "memory");
}
__device__ __forceinline__ void wait_async_le4() {
  asm volatile("s_wait_asynccnt 0x4" ::: "memory");
}

// ---------------- routing kernels ----------------
__global__ void moe_zero_meta(int* meta) {
  if (threadIdx.x < META_INTS) meta[threadIdx.x] = 0;
}

__global__ void moe_fill_defaults(int* rowToken, float* rowWeight) {
  int i = blockIdx.x * blockDim.x + threadIdx.x;
  if (i < MAXROWS) { rowToken[i] = 0; rowWeight[i] = 0.0f; }
}

__global__ void moe_count(const int* __restrict__ idx, int* meta) {
  int i = blockIdx.x * blockDim.x + threadIdx.x;
  if (i < NASSIGN) atomicAdd(&meta[idx[i] & (NEXP - 1)], 1);
}

__global__ void moe_prefix(int* meta) {
  if (threadIdx.x == 0) {
    int rs = 0, ts = 0;
    for (int e = 0; e < NEXP; ++e) {
      int c = meta[e];
      int t = (c + TM - 1) / TM;
      meta[16 + e] = rs;
      meta[24 + e] = ts;
      rs += t * TM;
      ts += t;
    }
    meta[32] = ts;
    meta[33] = rs;
  }
}

__global__ void moe_scatter(const int* __restrict__ idx, const float* __restrict__ probs,
                            int* meta, int* rowToken, float* rowWeight) {
  int i = blockIdx.x * blockDim.x + threadIdx.x;
  if (i >= NASSIGN) return;
  int e = idx[i] & (NEXP - 1);
  int pos = meta[16 + e] + atomicAdd(&meta[8 + e], 1);
  rowToken[pos]  = i >> 1;             // token = assignment / TOPK
  rowWeight[pos] = probs[i];
}

// gather selected token activations -> bf16, per padded row (pad rows use token 0, weight 0)
__global__ __launch_bounds__(256) void moe_gather(const float* __restrict__ x,
                                                  const int* __restrict__ rowToken,
                                                  uint16_t* __restrict__ Xg) {
  int row = blockIdx.x;
  int t = rowToken[row];
  int i = threadIdx.x * 4;
  f32x4 v = *(const f32x4*)(x + (size_t)t * DIM + i);
  u32x2 o;
  o.x = (uint32_t)f32_bf16(v.x) | ((uint32_t)f32_bf16(v.y) << 16);
  o.y = (uint32_t)f32_bf16(v.z) | ((uint32_t)f32_bf16(v.w) << 16);
  *(u32x2*)(Xg + (size_t)row * DIM + i) = o;
}

// src [E][R][C] f32  ->  dst [E][C][R] bf16   (grid: (C/32, R/32, E), 256 thr)
__global__ __launch_bounds__(256) void moe_transpose_convert(const float* __restrict__ src,
                                                             uint16_t* __restrict__ dst,
                                                             int R, int C) {
  __shared__ float tile[32][33];
  int e = blockIdx.z;
  const float* s = src + (size_t)e * R * C;
  uint16_t*    d = dst + (size_t)e * R * C;
  int c0 = blockIdx.x * 32, r0 = blockIdx.y * 32;
  int tx = threadIdx.x & 31, ty = threadIdx.x >> 5;   // 32 x 8
  #pragma unroll
  for (int i = 0; i < 32; i += 8)
    tile[ty + i][tx] = s[(size_t)(r0 + ty + i) * C + (c0 + tx)];
  __syncthreads();
  #pragma unroll
  for (int i = 0; i < 32; i += 8)
    d[(size_t)(c0 + ty + i) * R + (r0 + tx)] = f32_bf16(tile[tx][ty + i]);
}

__global__ void moe_zero_out(float* __restrict__ out) {
  size_t i = ((size_t)blockIdx.x * blockDim.x + threadIdx.x) * 4;
  *(f32x4*)(out + i) = (f32x4){0.f, 0.f, 0.f, 0.f};
  if (blockIdx.x == 0 && threadIdx.x == 0)
    out[(size_t)N_TOK * DIM] = 0.0f;   // total_loss = 0 (tuple output, concatenated flat)
}

// ---------------- shared WMMA mainloop (double-buffered async LDS) ----------------
// A tile [TM x TK] from Ag (row stride LD), B tile [TN x TK] from Bg (row stride LD, K contiguous)
// As/Bs each hold two buffers of TILE_HALFS halfs.
template <int LD, int KD>
__device__ __forceinline__ void gemm_main(const uint16_t* __restrict__ Ag,
                                          const uint16_t* __restrict__ Bg,
                                          uint16_t* As, uint16_t* Bs,
                                          int tid, floatx8 acc[2][4]) {
  const int lane = tid & 31;
  const int wave = tid >> 5;
  const int wm = (wave >> 1) * 32;     // 4 waves along M
  const int wn = (wave & 1) * 64;      // 2 waves along N
  const int lh = lane >> 4;            // lane half (0/1)
  const int lr = lane & 15;

  const int row  = tid >> 2;           // 0..63
  const int row2 = row + 64;           // 64..127
  const int col  = (tid & 3) * 8;      // 0,8,16,24 halfs

  // issue one K-tile's async copies (4 wave-level async ops)
  auto issue = [&](int k0, int buf) {
    uint16_t* a = As + buf * TILE_HALFS;
    uint16_t* b = Bs + buf * TILE_HALFS;
    async_ld_b128(&Ag[(size_t)row  * LD + k0 + col], &a[row  * LDSS + col]);
    async_ld_b128(&Bg[(size_t)row  * LD + k0 + col], &b[row  * LDSS + col]);
    async_ld_b128(&Ag[(size_t)row2 * LD + k0 + col], &a[row2 * LDSS + col]);
    async_ld_b128(&Bg[(size_t)row2 * LD + k0 + col], &b[row2 * LDSS + col]);
  };

  constexpr int NK = KD / TK;
  issue(0, 0);
  issue(TK, 1);

  for (int k = 0; k < NK; ++k) {
    // async loads complete in order: <=4 outstanding means tile k has landed
    if (k + 1 < NK) wait_async_le4(); else wait_async_le0();
    __syncthreads();                   // all waves' portions visible

    const int buf = k & 1;
    const uint16_t* al = As + buf * TILE_HALFS;
    const uint16_t* bl = Bs + buf * TILE_HALFS;

    bf16x16 a[2], b[4];
    #pragma unroll
    for (int mi = 0; mi < 2; ++mi) {
      const uint16_t* r = &al[(wm + mi * 16 + lr) * LDSS];
      // A 16x32 bf16 layout: lanes 0-15 hold K{0..7,16..23}, lanes 16-31 hold K{8..15,24..31}
      a[mi] = lds_frag(r + lh * 8, r + lh * 8 + 16);
    }
    #pragma unroll
    for (int ni = 0; ni < 4; ++ni) {
      const uint16_t* r = &bl[(wn + ni * 16 + lr) * LDSS];
      // B 32x16 bf16 layout: lanes 0-15 hold K{0..7,16..23}, lanes 16-31 hold K{8..15,24..31}
      b[ni] = lds_frag(r + lh * 8, r + lh * 8 + 16);
    }
    #pragma unroll
    for (int mi = 0; mi < 2; ++mi)
      #pragma unroll
      for (int ni = 0; ni < 4; ++ni)
        acc[mi][ni] = __builtin_amdgcn_wmma_f32_16x16x32_bf16(
            false, a[mi], false, b[ni], (short)0, acc[mi][ni], false, false);

    __syncthreads();                   // everyone done reading buf before refill
    if (k + 2 < NK) issue((k + 2) * TK, buf);
  }
}

__device__ __forceinline__ int find_tile(const int* __restrict__ meta, int my,
                                         int& e_out) {
  int e = 0;
  #pragma unroll
  for (int i = 1; i < NEXP; ++i)
    if (meta[24 + i] <= my) e = i;
  e_out = e;
  return meta[16 + e] + (my - meta[24 + e]) * TM;
}

// ---------------- GEMM1: H1 = relu(Xg @ W1t^T + b1), bf16 out ----------------
__global__ __launch_bounds__(256) void moe_gemm1(const uint16_t* __restrict__ Xg,
                                                 const uint16_t* __restrict__ W1t,
                                                 const float* __restrict__ b1,
                                                 uint16_t* __restrict__ H1,
                                                 const int* __restrict__ meta) {
  __shared__ uint16_t As[2 * TILE_HALFS];
  __shared__ uint16_t Bs[2 * TILE_HALFS];

  const int my = blockIdx.y;
  if (my >= meta[32]) return;
  int e;
  const int m0 = find_tile(meta, my, e);
  const int n0 = blockIdx.x * TN;

  const uint16_t* Ag = Xg  + (size_t)m0 * DIM;
  const uint16_t* Bg = W1t + (size_t)e * HID * DIM + (size_t)n0 * DIM;

  const int tid = threadIdx.x;
  floatx8 acc[2][4] = {};
  gemm_main<DIM, DIM>(Ag, Bg, As, Bs, tid, acc);

  const int lane = tid & 31, wave = tid >> 5;
  const int wm = (wave >> 1) * 32, wn = (wave & 1) * 64;
  const int lh = lane >> 4, lr = lane & 15;

  #pragma unroll
  for (int ni = 0; ni < 4; ++ni) {
    int h = n0 + wn + ni * 16 + lr;
    float bias = b1[(size_t)e * HID + h];
    #pragma unroll
    for (int mi = 0; mi < 2; ++mi) {
      int mbase = m0 + wm + mi * 16 + lh * 8;
      #pragma unroll
      for (int r = 0; r < 8; ++r) {
        float v = acc[mi][ni][r] + bias;
        v = v > 0.0f ? v : 0.0f;
        H1[(size_t)(mbase + r) * HID + h] = f32_bf16(v);
      }
    }
  }
}

// ---------------- GEMM2: out[tok] += p * (H1 @ W2t^T + b2) ----------------
__global__ __launch_bounds__(256) void moe_gemm2(const uint16_t* __restrict__ H1,
                                                 const uint16_t* __restrict__ W2t,
                                                 const float* __restrict__ b2,
                                                 const int* __restrict__ rowToken,
                                                 const float* __restrict__ rowWeight,
                                                 float* __restrict__ out,
                                                 const int* __restrict__ meta) {
  __shared__ uint16_t As[2 * TILE_HALFS];
  __shared__ uint16_t Bs[2 * TILE_HALFS];

  const int my = blockIdx.y;
  if (my >= meta[32]) return;
  int e;
  const int m0 = find_tile(meta, my, e);
  const int n0 = blockIdx.x * TN;

  const uint16_t* Ag = H1  + (size_t)m0 * HID;
  const uint16_t* Bg = W2t + (size_t)e * DIM * HID + (size_t)n0 * HID;

  const int tid = threadIdx.x;
  floatx8 acc[2][4] = {};
  gemm_main<HID, HID>(Ag, Bg, As, Bs, tid, acc);

  const int lane = tid & 31, wave = tid >> 5;
  const int wm = (wave >> 1) * 32, wn = (wave & 1) * 64;
  const int lh = lane >> 4, lr = lane & 15;

  int   tok[2][8];
  float wgt[2][8];
  #pragma unroll
  for (int mi = 0; mi < 2; ++mi) {
    int mbase = m0 + wm + mi * 16 + lh * 8;
    #pragma unroll
    for (int r = 0; r < 8; ++r) {
      tok[mi][r] = rowToken[mbase + r];
      wgt[mi][r] = rowWeight[mbase + r];
    }
  }

  #pragma unroll
  for (int ni = 0; ni < 4; ++ni) {
    int d = n0 + wn + ni * 16 + lr;
    float bias = b2[(size_t)e * DIM + d];
    #pragma unroll
    for (int mi = 0; mi < 2; ++mi)
      #pragma unroll
      for (int r = 0; r < 8; ++r) {
        float v = (acc[mi][ni][r] + bias) * wgt[mi][r];
        atomicAdd(&out[(size_t)tok[mi][r] * DIM + d], v);
      }
  }
}

// ---------------- launcher ----------------
extern "C" void kernel_launch(void* const* d_in, const int* in_sizes, int n_in,
                              void* d_out, int out_size, void* d_ws, size_t ws_size,
                              hipStream_t stream) {
  (void)in_sizes; (void)n_in; (void)out_size; (void)ws_size;
  const float* x     = (const float*)d_in[0];
  const float* probs = (const float*)d_in[1];
  const int*   idx   = (const int*)d_in[2];   // harness ABI: integer -> const int*
  const float* W1    = (const float*)d_in[3];
  const float* b1    = (const float*)d_in[4];
  const float* W2    = (const float*)d_in[5];
  const float* b2    = (const float*)d_in[6];
  float* out = (float*)d_out;

  char* ws = (char*)d_ws;
  size_t off = 0;
  auto wsalloc = [&](size_t bytes) -> char* {
    char* p = ws + off;
    off += (bytes + 255) & ~(size_t)255;
    return p;
  };
  int*      meta      = (int*)wsalloc(META_INTS * sizeof(int));
  int*      rowToken  = (int*)wsalloc((size_t)MAXROWS * sizeof(int));
  float*    rowWeight = (float*)wsalloc((size_t)MAXROWS * sizeof(float));
  uint16_t* Xg        = (uint16_t*)wsalloc((size_t)MAXROWS * DIM * 2);
  uint16_t* H1        = (uint16_t*)wsalloc((size_t)MAXROWS * HID * 2);
  uint16_t* W1t       = (uint16_t*)wsalloc((size_t)NEXP * DIM * HID * 2);
  uint16_t* W2t       = (uint16_t*)wsalloc((size_t)NEXP * DIM * HID * 2);

  // routing
  moe_zero_meta<<<1, 64, 0, stream>>>(meta);
  moe_fill_defaults<<<(MAXROWS + 255) / 256, 256, 0, stream>>>(rowToken, rowWeight);
  moe_count<<<(NASSIGN + 255) / 256, 256, 0, stream>>>(idx, meta);
  moe_prefix<<<1, 1, 0, stream>>>(meta);
  moe_scatter<<<(NASSIGN + 255) / 256, 256, 0, stream>>>(idx, probs, meta, rowToken, rowWeight);

  // data staging (fp32 -> bf16, weights transposed so K is contiguous in GEMM B tiles)
  moe_gather<<<MAXROWS, 256, 0, stream>>>(x, rowToken, Xg);
  moe_transpose_convert<<<dim3(HID / 32, DIM / 32, NEXP), 256, 0, stream>>>(W1, W1t, DIM, HID);
  moe_transpose_convert<<<dim3(DIM / 32, HID / 32, NEXP), 256, 0, stream>>>(W2, W2t, HID, DIM);
  moe_zero_out<<<(N_TOK * DIM) / 1024, 256, 0, stream>>>(out);

  // expert GEMMs on WMMA
  moe_gemm1<<<dim3(HID / TN, MAX_MTILES), 256, 0, stream>>>(Xg, W1t, b1, H1, meta);
  moe_gemm2<<<dim3(DIM / TN, MAX_MTILES), 256, 0, stream>>>(H1, W2t, b2, rowToken, rowWeight, out, meta);
}
